// SocialModel_6382321402280
// MI455X (gfx1250) — matrix operand order repaired and checked by reading, not compiled, not run on hardware
//
#include <hip/hip_runtime.h>

#define T_FRAMES 12
#define NPED     1000
#define KPAD     1024
#define NCHUNK   16   // KPAD / 64
#define GCELL    16
#define RNN      128
#define EMB      64
#define OUTD     5
#define GT_PITCH 72   // halves per row: 144B -> 16B aligned, conflict-free strides

typedef __attribute__((ext_vector_type(16))) __bf16 v16bf;
typedef __attribute__((ext_vector_type(8)))  __bf16 bf16x8;
typedef __attribute__((ext_vector_type(8)))  float  v8f;

__device__ __forceinline__ __bf16 tobf(float f) {
  // round-to-nearest-even f32 -> bf16
  unsigned u = __builtin_bit_cast(unsigned, f);
  unsigned r = (u + 0x7FFFu + ((u >> 16) & 1u)) >> 16;
  unsigned short s = (unsigned short)r;
  return __builtin_bit_cast(__bf16, s);
}

__device__ __forceinline__ v16bf cat8(bf16x8 lo, bf16x8 hi) {
  return __builtin_shufflevector(lo, hi, 0, 1, 2, 3, 4, 5, 6, 7,
                                         8, 9, 10, 11, 12, 13, 14, 15);
}

// ---------------------------------------------------------------------------
// prep: h/c init, pack [W_ih | W_hh] -> bf16 Wb[col][k], build initial hT
// hT is the bf16 TRANSPOSED hidden state [j=128][p=1024], zero-padded p>=1000.
// ---------------------------------------------------------------------------
__global__ __launch_bounds__(256) void prep_kernel(
    const float* __restrict__ hs, const float* __restrict__ cs,
    const float* __restrict__ W_ih, const float* __restrict__ W_hh,
    float* __restrict__ h, float* __restrict__ c,
    __bf16* __restrict__ Wb, __bf16* __restrict__ hT) {
  int i = blockIdx.x * blockDim.x + threadIdx.x;
  if (i < NPED * RNN) { h[i] = hs[i]; c[i] = cs[i]; }
  if (i < 512 * 256) {
    int col = i >> 8, k = i & 255;
    float v = (k < 128) ? W_ih[col * 128 + k] : W_hh[col * 128 + (k - 128)];
    Wb[i] = tobf(v);
  }
  if (i < RNN * KPAD) {
    int j = i >> 10, p = i & (KPAD - 1);
    float v = (p < NPED) ? hs[(size_t)p * RNN + j] : 0.f;
    hT[i] = tobf(v);
  }
}

// ---------------------------------------------------------------------------
// social kernel helpers
// ---------------------------------------------------------------------------
// stage one 64p x 16g grid chunk, transposed, into an LDS buffer
__device__ __forceinline__ void stage_grid(const float* __restrict__ g,
                                           __bf16* __restrict__ dst,
                                           int k0, int sp, int sq, bool tail) {
  float4 a, b;
  if (!tail) {
    a = *(const float4*)(g + (size_t)(k0 + sp) * GCELL + sq);
    b = *(const float4*)(g + (size_t)(k0 + sp + 32) * GCELL + sq);
  } else {
    int p0 = k0 + sp, p1 = k0 + sp + 32;
    a = (p0 < NPED) ? *(const float4*)(g + (size_t)p0 * GCELL + sq)
                    : make_float4(0.f, 0.f, 0.f, 0.f);
    b = (p1 < NPED) ? *(const float4*)(g + (size_t)p1 * GCELL + sq)
                    : make_float4(0.f, 0.f, 0.f, 0.f);
  }
  dst[(sq + 0) * GT_PITCH + sp] = tobf(a.x);
  dst[(sq + 1) * GT_PITCH + sp] = tobf(a.y);
  dst[(sq + 2) * GT_PITCH + sp] = tobf(a.z);
  dst[(sq + 3) * GT_PITCH + sp] = tobf(a.w);
  dst[(sq + 0) * GT_PITCH + sp + 32] = tobf(b.x);
  dst[(sq + 1) * GT_PITCH + sp + 32] = tobf(b.y);
  dst[(sq + 2) * GT_PITCH + sp + 32] = tobf(b.z);
  dst[(sq + 3) * GT_PITCH + sp + 32] = tobf(b.w);
}

// consume one 64-K chunk: 2 A fragments from LDS, 4 B fragments from hT, 4 WMMA
__device__ __forceinline__ void social_compute(const __bf16* __restrict__ buf,
                                               const __bf16* __restrict__ hT,
                                               int k0, int m, int kbA, int kbB,
                                               int jt0, int jt1,
                                               v8f& acc0, v8f& acc1) {
#pragma unroll
  for (int kc = 0; kc < 64; kc += 32) {
    bf16x8 alo = *(const bf16x8*)&buf[m * GT_PITCH + kc + kbA];
    bf16x8 ahi = *(const bf16x8*)&buf[m * GT_PITCH + kc + kbA + 8];
    v16bf a = cat8(alo, ahi);
    const __bf16* hp0 = hT + (size_t)jt0 * KPAD + k0 + kc + kbB;
    const __bf16* hp1 = hT + (size_t)jt1 * KPAD + k0 + kc + kbB;
    v16bf b0 = cat8(*(const bf16x8*)hp0, *(const bf16x8*)(hp0 + 8));
    v16bf b1 = cat8(*(const bf16x8*)hp1, *(const bf16x8*)(hp1 + 8));
    acc0 = __builtin_amdgcn_wmma_f32_16x16x32_bf16(false, a, false, b0, (short)0, acc0, false, false);
    acc1 = __builtin_amdgcn_wmma_f32_16x16x32_bf16(false, a, false, b1, (short)0, acc1, false, false);
  }
}

// ---------------------------------------------------------------------------
// social pooling + W_soc projection, fused.  One block (128 thr) per ped n.
// S = grid[n]^T @ h  (M=16 cells, Ncols=128 hidden, K=1024 padded peds)
// Double-buffered LDS grid tiles, branch-free main loop, peeled tail.
// te[n] = relu(W_soc @ vec(S) + b_soc)
// ---------------------------------------------------------------------------
__global__ __launch_bounds__(128) void social_te_kernel(
    const float* __restrict__ grid_t,   // [N][N][16] for this frame (f32)
    const __bf16* __restrict__ hT,      // [128][1024] bf16 transposed hidden
    const float* __restrict__ W_soc,    // [64][2048]
    const float* __restrict__ b_soc,    // [64]
    float* __restrict__ te_out) {       // [N][64]
  const int n = blockIdx.x;
  const float* __restrict__ g = grid_t + (size_t)n * NPED * GCELL;

  __shared__ __align__(16) __bf16 sGT[2][GCELL * GT_PITCH];  // double buffer
  __shared__ __align__(16) float  sS[GCELL * RNN];
  __shared__ float red[128];

  const int tid  = threadIdx.x;
  const int lane = tid & 31;
  const int wave = tid >> 5;

  const int sp = tid >> 2;              // staging row p (0..31)
  const int sq = (tid & 3) * 4;         // staging col g base

  const int m   = lane & 15;            // A row (grid cell)
  const int kbA = (lane >> 4) * 16;     // A K-base within 32-chunk
  const int kbB = (lane >> 4) * 16;     // B K-base within 32-chunk
  const int jt0 = wave * 32 + (lane & 15);
  const int jt1 = jt0 + 16;

  v8f acc0 = {}; v8f acc1 = {};

  // prologue: stage chunk 0
  stage_grid(g, sGT[0], 0, sp, sq, false);
  __syncthreads();

  // hot loop: chunks 0..13, stage i+1 (unguarded: chunk 14 ends at p=959<1000)
  for (int it = 0; it < NCHUNK - 2; ++it) {
    const int k0 = it * 64;
    stage_grid(g, sGT[(it + 1) & 1], k0 + 64, sp, sq, false);
    if (it + 2 < NCHUNK)
      __builtin_prefetch(g + (size_t)(k0 + 128 + sp) * GCELL + sq, 0, 0);
    social_compute(sGT[it & 1], hT, k0, m, kbA, kbB, jt0, jt1, acc0, acc1);
    __syncthreads();
  }
  // it = 14: stage guarded tail chunk 15 (p 960..1023, valid < 1000)
  stage_grid(g, sGT[(NCHUNK - 1) & 1], (NCHUNK - 1) * 64, sp, sq, true);
  social_compute(sGT[(NCHUNK - 2) & 1], hT, (NCHUNK - 2) * 64,
                 m, kbA, kbB, jt0, jt1, acc0, acc1);
  __syncthreads();
  // it = 15: compute only
  social_compute(sGT[(NCHUNK - 1) & 1], hT, (NCHUNK - 1) * 64,
                 m, kbA, kbB, jt0, jt1, acc0, acc1);

  // C/D layout: VGPR r, lanes 0-15 -> M=r, N=lane; lanes 16-31 -> M=r+8
  {
    int crow = (lane >> 4) * 8;
    int ccol = lane & 15;
#pragma unroll
    for (int r = 0; r < 8; ++r) {
      sS[(crow + r) * RNN + wave * 32 + ccol]      = acc0[r];
      sS[(crow + r) * RNN + wave * 32 + 16 + ccol] = acc1[r];
    }
  }
  __syncthreads();

  // te = relu(W_soc @ vec(S) + b_soc); vec idx = g*128 + j == sS flat layout
  {
    int j = tid & 63, halfk = tid >> 6;
    const float4* wrow = (const float4*)(W_soc + (size_t)j * (GCELL * RNN) + halfk * 1024);
    const float4* svec = (const float4*)(sS + halfk * 1024);
    float sum = 0.f;
#pragma unroll 4
    for (int k = 0; k < 256; ++k) {
      float4 w = wrow[k], s = svec[k];
      sum += w.x * s.x + w.y * s.y + w.z * s.z + w.w * s.w;
    }
    red[tid] = sum;
  }
  __syncthreads();
  if (tid < 64) {
    float r = red[tid] + red[tid + 64] + b_soc[tid];
    te_out[(size_t)n * EMB + tid] = fmaxf(r, 0.f);
  }
}

// ---------------------------------------------------------------------------
// LSTM step: blocks of 16 pedestrians.
// gates = [ie | te | h] @ Wb^T  (M=16, N=512, K=256) via WMMA, then pointwise.
// Also emits bf16 transposed hT for the next frame's social kernel.
// ---------------------------------------------------------------------------
__global__ __launch_bounds__(128) void lstm_step_kernel(
    const float* __restrict__ x_t,    // [N][2]
    const float* __restrict__ te,     // [N][64]
    const float* __restrict__ W_in,   // [64][2]
    const float* __restrict__ b_in,   // [64]
    const __bf16* __restrict__ Wb,    // [512][256] bf16 packed
    const float* __restrict__ b_ih,   // [512]
    const float* __restrict__ b_hh,   // [512]
    const float* __restrict__ W_out,  // [5][128]
    const float* __restrict__ b_out,  // [5]
    float* __restrict__ h,            // [N][128] in/out (f32)
    float* __restrict__ c,            // [N][128] in/out (f32)
    __bf16* __restrict__ hT,          // [128][1024] out (bf16 transposed)
    float* __restrict__ out_t) {      // [N][5]
  const int ptile = blockIdx.x;       // peds ptile*16 .. +15
  const int tid   = threadIdx.x;
  const int lane  = tid & 31;
  const int wave  = tid >> 5;

  __shared__ __align__(16) __bf16 sA[16 * 256];   // [ped][k]: ie | te | h
  __shared__ float  sGate[16 * 512];
  __shared__ float  sHC[16 * RNN];

  // ie = relu(x @ W_in^T + b_in)
  for (int i = tid; i < 16 * 64; i += 128) {
    int pl = i >> 6, jj = i & 63;
    int ped = ptile * 16 + pl;
    float v = 0.f;
    if (ped < NPED) {
      float x0 = x_t[ped * 2], x1 = x_t[ped * 2 + 1];
      v = fmaxf(W_in[jj * 2] * x0 + W_in[jj * 2 + 1] * x1 + b_in[jj], 0.f);
    }
    sA[pl * 256 + jj] = tobf(v);
  }
  for (int i = tid; i < 16 * 64; i += 128) {
    int pl = i >> 6, jj = i & 63;
    int ped = ptile * 16 + pl;
    float v = (ped < NPED) ? te[(size_t)ped * 64 + jj] : 0.f;
    sA[pl * 256 + 64 + jj] = tobf(v);
  }
  for (int i = tid; i < 16 * 128; i += 128) {
    int pl = i >> 7, jj = i & 127;
    int ped = ptile * 16 + pl;
    float v = (ped < NPED) ? h[(size_t)ped * 128 + jj] : 0.f;
    sA[pl * 256 + 128 + jj] = tobf(v);
  }
  __syncthreads();

  // gate GEMM: 32 N-tiles, each wave owns 8 of them
  v8f acc[8];
#pragma unroll
  for (int t = 0; t < 8; ++t) acc[t] = (v8f){};
  const int m  = lane & 15;
  const int kb = (lane >> 4) * 16;
  for (int k0 = 0; k0 < 256; k0 += 32) {
    bf16x8 alo = *(const bf16x8*)&sA[m * 256 + k0 + kb];
    bf16x8 ahi = *(const bf16x8*)&sA[m * 256 + k0 + kb + 8];
    v16bf a = cat8(alo, ahi);
#pragma unroll
    for (int t = 0; t < 8; ++t) {
      int ncol = (t * 4 + wave) * 16 + (lane & 15);
      const __bf16* wp = Wb + (size_t)ncol * 256 + k0 + (lane >> 4) * 16;
      v16bf b = cat8(*(const bf16x8*)wp, *(const bf16x8*)(wp + 8));
      acc[t] = __builtin_amdgcn_wmma_f32_16x16x32_bf16(false, a, false, b, (short)0, acc[t], false, false);
    }
  }
  {
    int crow = (lane >> 4) * 8;
    int ccol = lane & 15;
#pragma unroll
    for (int t = 0; t < 8; ++t) {
      int colb = (t * 4 + wave) * 16 + ccol;
#pragma unroll
      for (int r = 0; r < 8; ++r) sGate[(crow + r) * 512 + colb] = acc[t][r];
    }
  }
  __syncthreads();

  // pointwise LSTM (gate order i, f, g, o); also emit bf16 transposed hT
  for (int i = tid; i < 16 * 128; i += 128) {
    int pl = i >> 7, jj = i & 127;
    int ped = ptile * 16 + pl;
    float hn = 0.f;
    if (ped < NPED) {
      float gi = sGate[pl * 512 + jj]       + b_ih[jj]       + b_hh[jj];
      float gf = sGate[pl * 512 + 128 + jj] + b_ih[128 + jj] + b_hh[128 + jj];
      float gg = sGate[pl * 512 + 256 + jj] + b_ih[256 + jj] + b_hh[256 + jj];
      float go = sGate[pl * 512 + 384 + jj] + b_ih[384 + jj] + b_hh[384 + jj];
      float si = 1.f / (1.f + expf(-gi));
      float sf = 1.f / (1.f + expf(-gf));
      float so = 1.f / (1.f + expf(-go));
      float cn = sf * c[(size_t)ped * 128 + jj] + si * tanhf(gg);
      hn = so * tanhf(cn);
      c[(size_t)ped * 128 + jj] = cn;
      h[(size_t)ped * 128 + jj] = hn;
      hT[(size_t)jj * KPAD + ped] = tobf(hn);
    }
    sHC[pl * 128 + jj] = hn;
  }
  __syncthreads();

  // output projection: 16 peds x 5 outputs
  if (tid < 16 * OUTD) {
    int pl = tid / OUTD, oj = tid % OUTD;
    int ped = ptile * 16 + pl;
    if (ped < NPED) {
      float s = b_out[oj];
#pragma unroll 4
      for (int k = 0; k < 128; ++k) s += W_out[oj * 128 + k] * sHC[pl * 128 + k];
      out_t[(size_t)ped * OUTD + oj] = s;
    }
  }
}

// ---------------------------------------------------------------------------
extern "C" void kernel_launch(void* const* d_in, const int* in_sizes, int n_in,
                              void* d_out, int out_size, void* d_ws, size_t ws_size,
                              hipStream_t stream) {
  const float* input_data = (const float*)d_in[0];   // [T][N][2]
  const float* grids      = (const float*)d_in[1];   // [T][N][N][16]
  const float* hidden0    = (const float*)d_in[2];   // [N][128]
  const float* cell0      = (const float*)d_in[3];   // [N][128]
  const float* W_in       = (const float*)d_in[4];   // [64][2]
  const float* b_in       = (const float*)d_in[5];   // [64]
  const float* W_soc      = (const float*)d_in[6];   // [64][2048]
  const float* b_soc      = (const float*)d_in[7];   // [64]
  const float* W_ih       = (const float*)d_in[8];   // [512][128]
  const float* W_hh       = (const float*)d_in[9];   // [512][128]
  const float* b_ih       = (const float*)d_in[10];  // [512]
  const float* b_hh       = (const float*)d_in[11];  // [512]
  const float* W_out      = (const float*)d_in[12];  // [5][128]
  const float* b_out      = (const float*)d_in[13];  // [5]
  float* out = (float*)d_out;

  // workspace layout
  float*  h_ws  = (float*)d_ws;                   // N*128 f32
  float*  c_ws  = h_ws + NPED * RNN;              // N*128 f32
  float*  te_ws = c_ws + NPED * RNN;              // N*64  f32
  __bf16* Wb    = (__bf16*)(te_ws + NPED * EMB);  // 512*256 bf16
  __bf16* hT    = Wb + 512 * 256;                 // 128*1024 bf16

  prep_kernel<<<512, 256, 0, stream>>>(hidden0, cell0, W_ih, W_hh,
                                       h_ws, c_ws, Wb, hT);

  const int lstm_blocks = (NPED + 15) / 16;       // 63
  for (int t = 0; t < T_FRAMES; ++t) {
    social_te_kernel<<<NPED, 128, 0, stream>>>(
        grids + (size_t)t * NPED * NPED * GCELL, hT, W_soc, b_soc, te_ws);
    lstm_step_kernel<<<lstm_blocks, 128, 0, stream>>>(
        input_data + (size_t)t * NPED * 2, te_ws, W_in, b_in, Wb,
        b_ih, b_hh, W_out, b_out, h_ws, c_ws, hT,
        out + (size_t)t * NPED * OUTD);
  }

  // append final h, c to outputs (reference returns (outputs, h, c))
  hipMemcpyAsync(out + (size_t)T_FRAMES * NPED * OUTD, h_ws,
                 (size_t)NPED * RNN * sizeof(float), hipMemcpyDeviceToDevice, stream);
  hipMemcpyAsync(out + (size_t)T_FRAMES * NPED * OUTD + NPED * RNN, c_ws,
                 (size_t)NPED * RNN * sizeof(float), hipMemcpyDeviceToDevice, stream);
}
